// LlamaAttentionTP_33998961115937
// MI455X (gfx1250) — matrix operand order repaired and checked by reading, not compiled, not run on hardware
//
#include <hip/hip_runtime.h>
#include <hip/hip_bf16.h>
#include <math.h>

#define BATCH  2
#define SEQ    2048
#define HIDDEN 2048
#define HEADS  16
#define DK     128
#define BS     (BATCH*SEQ)   // 4096 rows

typedef unsigned short u16;
typedef unsigned int   u32;
typedef __attribute__((ext_vector_type(16))) __bf16 v16bf;
typedef __attribute__((ext_vector_type(8)))  float  v8f;
typedef __attribute__((ext_vector_type(4)))  int    v4i;

union Frag16 { v16bf v; uint4 q[2]; u16 h[16]; };

#if __has_builtin(__builtin_amdgcn_global_load_async_to_lds_b128)
#define HAVE_ASYNC_LDS 1
#else
#define HAVE_ASYNC_LDS 0
#endif

// pointee type for the async builtin is int4 with explicit address spaces
typedef __attribute__((address_space(1))) v4i gv4i;   // global
typedef __attribute__((address_space(3))) v4i lv4i;   // LDS

__device__ __forceinline__ u16 f2b(float f) {
  u32 u = __float_as_uint(f);
  u += 0x7FFFu + ((u >> 16) & 1u);       // round-to-nearest-even
  return (u16)(u >> 16);
}
__device__ __forceinline__ float b2f(u16 h) { return __uint_as_float(((u32)h) << 16); }

__device__ __forceinline__ v8f v8f_zero() {
  v8f z = {0.f,0.f,0.f,0.f,0.f,0.f,0.f,0.f};
  return z;
}
__device__ __forceinline__ v8f wmma_bf16(v16bf a, v16bf b, v8f c) {
  return __builtin_amdgcn_wmma_f32_16x16x32_bf16(false, a, false, b, (short)0, c, false, false);
}

// stage 16 bf16 (two b128 beats) global -> LDS; async (VGPR-bypassing) on CDNA5
__device__ __forceinline__ void stage_b256(const u16* g, u16* l) {
#if HAVE_ASYNC_LDS
  u16* gnc = const_cast<u16*>(g);
  __builtin_amdgcn_global_load_async_to_lds_b128((gv4i*)gnc,       (lv4i*)l,       0, 0);
  __builtin_amdgcn_global_load_async_to_lds_b128((gv4i*)(gnc + 8), (lv4i*)(l + 8), 0, 0);
#else
  uint4 t0 = *(const uint4*)g;
  uint4 t1 = *(const uint4*)(g + 8);
  *(uint4*)l       = t0;
  *(uint4*)(l + 8) = t1;
#endif
}
__device__ __forceinline__ void async_wait_all() {
#if HAVE_ASYNC_LDS
#if __has_builtin(__builtin_amdgcn_s_wait_asynccnt)
  __builtin_amdgcn_s_wait_asynccnt(0);
#else
  asm volatile("s_wait_asynccnt 0x0" ::: "memory");
#endif
#endif
}

// ---------------------------------------------------------------------------
// 1) f32 -> bf16 downcast
// ---------------------------------------------------------------------------
__global__ __launch_bounds__(256) void f32_to_bf16_kernel(const float* __restrict__ in,
                                                          u16* __restrict__ out, int n) {
  int i = blockIdx.x * 256 + threadIdx.x;
  if (i < n) out[i] = f2b(in[i]);
}

// ---------------------------------------------------------------------------
// 2) C[M,N] = A[M,K] * W[N,K]^T   (A,W bf16; out bf16 or f32)
//    128x128 block tile, 8 waves, wave tile 64x32 (4x2 WMMA tiles), K step 32.
//    Double-buffered LDS; next panel staged asynchronously during compute.
// ---------------------------------------------------------------------------
__global__ __launch_bounds__(256) void gemm_xwt_kernel(const u16* __restrict__ A,
                                                       const u16* __restrict__ W,
                                                       void* __restrict__ Cout,
                                                       int M, int N, int K, int out_f32) {
  __shared__ u16 As[2][128 * 40];   // padded stride 40 elems (80B, 16B-multiple)
  __shared__ u16 Ws[2][128 * 40];

  const int tid  = threadIdx.x;
  const int lane = tid & 31, wave = tid >> 5;
  const int wm = wave >> 2, wn = wave & 3;          // 2 x 4 wave grid
  const int bm = blockIdx.y * 128, bn = blockIdx.x * 128;
  const int lrow = lane & 15, khalf = lane >> 4;
  const int ar = tid >> 1, ac = tid & 1;            // 128 rows x 2 half-rows of 16

  const u16* gA = A + (size_t)(bm + ar) * K + ac * 16;
  const u16* gW = W + (size_t)(bn + ar) * K + ac * 16;
  u16* lA[2] = { &As[0][ar * 40 + ac * 16], &As[1][ar * 40 + ac * 16] };
  u16* lW[2] = { &Ws[0][ar * 40 + ac * 16], &Ws[1][ar * 40 + ac * 16] };

  v8f acc[4][2];
#pragma unroll
  for (int i = 0; i < 4; ++i)
#pragma unroll
    for (int j = 0; j < 2; ++j) acc[i][j] = v8f_zero();

  // prologue: stage first K panel into buffer 0
  stage_b256(gA, lA[0]);
  stage_b256(gW, lW[0]);

  for (int k0 = 0; k0 < K; k0 += 32) {
    const int cur = (k0 >> 5) & 1;
    async_wait_all();          // our staged data for `cur` has landed
    __syncthreads();           // everyone's has; prior readers of `cur^1` done

    if (k0 + 32 < K) {         // fire-and-forget next panel into other buffer
      stage_b256(gA + k0 + 32, lA[cur ^ 1]);
      stage_b256(gW + k0 + 32, lW[cur ^ 1]);
      if (k0 + 64 < K) {
        __builtin_prefetch(gA + k0 + 64, 0, 1);
        __builtin_prefetch(gW + k0 + 64, 0, 1);
      }
    }

    Frag16 af[4], bf[2];
#pragma unroll
    for (int mt = 0; mt < 4; ++mt) {   // A 16x32: lane=M, elems 0-7 -> K=khalf*8.., 8-15 -> 16+khalf*8..
      const u16* p = &As[cur][(wm * 64 + mt * 16 + lrow) * 40];
      af[mt].q[0] = *(const uint4*)(p + khalf * 8);
      af[mt].q[1] = *(const uint4*)(p + 16 + khalf * 8);
    }
#pragma unroll
    for (int nt = 0; nt < 2; ++nt) {   // B 32x16: lane=N, elems 0-7 -> K=khalf*8.., 8-15 -> 16+khalf*8..
      const u16* p = &Ws[cur][(wn * 32 + nt * 16 + lrow) * 40];
      bf[nt].q[0] = *(const uint4*)(p + khalf * 8);
      bf[nt].q[1] = *(const uint4*)(p + 16 + khalf * 8);
    }
#pragma unroll
    for (int mt = 0; mt < 4; ++mt)
#pragma unroll
      for (int nt = 0; nt < 2; ++nt)
        acc[mt][nt] = wmma_bf16(af[mt].v, bf[nt].v, acc[mt][nt]);
  }

  // C layout: lane = N (+16-lane half selects M+8), VGPR r -> M = r + 8*khalf
#pragma unroll
  for (int mt = 0; mt < 4; ++mt)
#pragma unroll
    for (int nt = 0; nt < 2; ++nt)
#pragma unroll
      for (int r = 0; r < 8; ++r) {
        int row = bm + wm * 64 + mt * 16 + r + khalf * 8;
        int col = bn + wn * 32 + nt * 16 + lrow;
        float v = acc[mt][nt][r];
        if (out_f32) ((float*)Cout)[(size_t)row * N + col] = v;
        else         ((u16*)Cout)[(size_t)row * N + col]   = f2b(v);
      }
}

// ---------------------------------------------------------------------------
// 3) RoPE in place on Q and K (bf16), freqs_cis f32 [SEQ][64][2]
// ---------------------------------------------------------------------------
__global__ __launch_bounds__(256) void rope_kernel(u16* __restrict__ q, u16* __restrict__ k,
                                                   const float* __restrict__ fc) {
  int pid = blockIdx.x * 256 + threadIdx.x;   // BS * 1024 pairs
  int row = pid >> 10;
  int cp  = pid & 1023;
  int s   = row & (SEQ - 1);
  int i   = cp & 63;                          // pair index within head
  float c  = fc[(s * 64 + i) * 2 + 0];
  float sn = fc[(s * 64 + i) * 2 + 1];
  size_t base = (size_t)row * HIDDEN + cp * 2;
  float q0 = b2f(q[base]), q1 = b2f(q[base + 1]);
  q[base]     = f2b(q0 * c - q1 * sn);
  q[base + 1] = f2b(q0 * sn + q1 * c);
  float k0 = b2f(k[base]), k1 = b2f(k[base + 1]);
  k[base]     = f2b(k0 * c - k1 * sn);
  k[base + 1] = f2b(k0 * sn + k1 * c);
}

// ---------------------------------------------------------------------------
// 4) Causal flash attention. grid = (SEQ/128, BATCH*HEADS), 256 threads.
//    wave w owns 16 query rows; 32-key steps; online softmax; WMMA everywhere.
// ---------------------------------------------------------------------------
__global__ __launch_bounds__(256) void flash_kernel(const u16* __restrict__ Q,
                                                    const u16* __restrict__ Kc,
                                                    const u16* __restrict__ V,
                                                    u16* __restrict__ Ctx) {
  __shared__ u16 Ks[32 * 136];       // keys x d, stride 136 (272B)
  __shared__ u16 Vt[128 * 40];       // d x keys (transposed), stride 40
  __shared__ u16 Ps[8][16 * 40];     // per-wave P scratch

  const int tid = threadIdx.x, lane = tid & 31, wave = tid >> 5;
  const int lrow = lane & 15, khalf = lane >> 4;
  const int qblk = blockIdx.x;
  const int bh = blockIdx.y, b = bh >> 4, h = bh & 15;
  const float scale = 0.08838834764831845f;   // 1/sqrt(128)

  // Q fragments: A 16x32 per 32-wide d chunk, loaded straight from global
  Frag16 qf[4];
  const size_t qrow = (size_t)(b * SEQ + qblk * 128 + wave * 16 + lrow);
#pragma unroll
  for (int kc = 0; kc < 4; ++kc) {
    const u16* p = Q + qrow * HIDDEN + h * 128 + kc * 32;
    qf[kc].q[0] = *(const uint4*)(p + khalf * 8);
    qf[kc].q[1] = *(const uint4*)(p + 16 + khalf * 8);
  }

  v8f o[8];
#pragma unroll
  for (int i = 0; i < 8; ++i) o[i] = v8f_zero();
  float mrow[8], lsum[8];
#pragma unroll
  for (int r = 0; r < 8; ++r) { mrow[r] = -3.0e38f; lsum[r] = 0.f; }

  const int nj = (qblk + 1) * 4;              // causal: only blocks <= q range
  const int kr = tid >> 3, cc = tid & 7;      // 32 rows x 8 col-chunks of 16

  for (int j = 0; j < nj; ++j) {
    const int jb = j * 32;
    __syncthreads();                          // previous Ks/Vt fully consumed
    {   // stage K (async, row-major) and V (transposed via VGPR scatter)
      const u16* g = Kc + (size_t)(b * SEQ + jb + kr) * HIDDEN + h * 128 + cc * 16;
      stage_b256(g, &Ks[kr * 136 + cc * 16]);
      const u16* gv = V + (size_t)(b * SEQ + jb + kr) * HIDDEN + h * 128 + cc * 16;
      Frag16 tmp;
      tmp.q[0] = *(const uint4*)gv;
      tmp.q[1] = *(const uint4*)(gv + 8);
#pragma unroll
      for (int e = 0; e < 16; ++e) Vt[(cc * 16 + e) * 40 + kr] = tmp.h[e];
    }
    async_wait_all();
    __syncthreads();

    // S = Q @ K^T : two 16x16 tiles over 32 keys
    v8f sc0 = v8f_zero(), sc1 = v8f_zero();
#pragma unroll
    for (int kc = 0; kc < 4; ++kc) {
      Frag16 kb0, kb1;
      const u16* p0 = &Ks[lrow * 136 + kc * 32];
      kb0.q[0] = *(const uint4*)(p0 + khalf * 8); kb0.q[1] = *(const uint4*)(p0 + 16 + khalf * 8);
      const u16* p1 = &Ks[(16 + lrow) * 136 + kc * 32];
      kb1.q[0] = *(const uint4*)(p1 + khalf * 8); kb1.q[1] = *(const uint4*)(p1 + 16 + khalf * 8);
      sc0 = wmma_bf16(qf[kc].v, kb0.v, sc0);
      sc1 = wmma_bf16(qf[kc].v, kb1.v, sc1);
    }

    // online softmax (rows live in 16-lane halves of the C layout)
    const int qg = qblk * 128 + wave * 16 + khalf * 8;
    float p0a[8], p1a[8];
#pragma unroll
    for (int r = 0; r < 8; ++r) {
      int qi = qg + r;
      float s0 = (jb + lrow      <= qi) ? sc0[r] * scale : -3.0e38f;
      float s1 = (jb + 16 + lrow <= qi) ? sc1[r] * scale : -3.0e38f;
      float mt = fmaxf(s0, s1);
#pragma unroll
      for (int off = 1; off < 16; off <<= 1) mt = fmaxf(mt, __shfl_xor(mt, off, 16));
      float mnew  = fmaxf(mrow[r], mt);
      float alpha = __expf(mrow[r] - mnew);
      float p0 = __expf(s0 - mnew), p1 = __expf(s1 - mnew);
      float rs = p0 + p1;
#pragma unroll
      for (int off = 1; off < 16; off <<= 1) rs += __shfl_xor(rs, off, 16);
      lsum[r] = lsum[r] * alpha + rs;
      mrow[r] = mnew;
#pragma unroll
      for (int t = 0; t < 8; ++t) o[t][r] *= alpha;
      p0a[r] = p0; p1a[r] = p1;
    }

    // relayout P (C layout -> A fragment) through per-wave LDS
    u16* pw = &Ps[wave][0];
#pragma unroll
    for (int r = 0; r < 8; ++r) {
      pw[(r + khalf * 8) * 40 + lrow]      = f2b(p0a[r]);
      pw[(r + khalf * 8) * 40 + 16 + lrow] = f2b(p1a[r]);
    }
    asm volatile("s_wait_dscnt 0" ::: "memory");   // same-wave LDS RAW
    Frag16 pf;
    {
      const u16* p = &Ps[wave][lrow * 40];
      pf.q[0] = *(const uint4*)(p + khalf * 8);
      pf.q[1] = *(const uint4*)(p + 16 + khalf * 8);
    }

    // O += P @ V : 8 d-tiles, B fragments contiguous thanks to transposed Vt
#pragma unroll
    for (int nt = 0; nt < 8; ++nt) {
      Frag16 vb;
      const u16* p = &Vt[(nt * 16 + lrow) * 40];
      vb.q[0] = *(const uint4*)(p + khalf * 8);
      vb.q[1] = *(const uint4*)(p + 16 + khalf * 8);
      o[nt] = wmma_bf16(pf.v, vb.v, o[nt]);
    }
  }

  // epilogue: O /= l, write bf16 context
#pragma unroll
  for (int r = 0; r < 8; ++r) {
    float inv = 1.0f / lsum[r];
    size_t row = (size_t)(b * SEQ + qblk * 128 + wave * 16 + r + khalf * 8);
#pragma unroll
    for (int nt = 0; nt < 8; ++nt)
      Ctx[row * HIDDEN + h * 128 + nt * 16 + lrow] = f2b(o[nt][r] * inv);
  }
}

// ---------------------------------------------------------------------------
// host
// ---------------------------------------------------------------------------
extern "C" void kernel_launch(void* const* d_in, const int* in_sizes, int n_in,
                              void* d_out, int out_size, void* d_ws, size_t ws_size,
                              hipStream_t stream) {
  (void)in_sizes; (void)n_in; (void)out_size; (void)ws_size;
  const float* x  = (const float*)d_in[0];
  const float* fc = (const float*)d_in[1];
  const float* wq = (const float*)d_in[2];
  const float* wk = (const float*)d_in[3];
  const float* wv = (const float*)d_in[4];
  const float* wo = (const float*)d_in[5];

  char* ws = (char*)d_ws;
  size_t off = 0;
  auto alloc = [&](size_t bytes) -> void* {
    void* p = ws + off;
    off += (bytes + 255) & ~(size_t)255;
    return p;
  };
  u16* xb  = (u16*)alloc((size_t)BS * HIDDEN * 2);
  u16* wqb = (u16*)alloc((size_t)HIDDEN * HIDDEN * 2);
  u16* wkb = (u16*)alloc((size_t)HIDDEN * HIDDEN * 2);
  u16* wvb = (u16*)alloc((size_t)HIDDEN * HIDDEN * 2);
  u16* wob = (u16*)alloc((size_t)HIDDEN * HIDDEN * 2);
  u16* qb  = (u16*)alloc((size_t)BS * HIDDEN * 2);
  u16* kb  = (u16*)alloc((size_t)BS * HIDDEN * 2);
  u16* vb  = (u16*)alloc((size_t)BS * HIDDEN * 2);
  u16* ctx = (u16*)alloc((size_t)BS * HIDDEN * 2);

  const int nx = BS * HIDDEN;        // 8,388,608
  const int nw = HIDDEN * HIDDEN;    // 4,194,304
  f32_to_bf16_kernel<<<nx / 256, 256, 0, stream>>>(x,  xb,  nx);
  f32_to_bf16_kernel<<<nw / 256, 256, 0, stream>>>(wq, wqb, nw);
  f32_to_bf16_kernel<<<nw / 256, 256, 0, stream>>>(wk, wkb, nw);
  f32_to_bf16_kernel<<<nw / 256, 256, 0, stream>>>(wv, wvb, nw);
  f32_to_bf16_kernel<<<nw / 256, 256, 0, stream>>>(wo, wob, nw);

  dim3 gg(HIDDEN / 128, BS / 128);
  gemm_xwt_kernel<<<gg, 256, 0, stream>>>(xb, wqb, qb, BS, HIDDEN, HIDDEN, 0);
  gemm_xwt_kernel<<<gg, 256, 0, stream>>>(xb, wkb, kb, BS, HIDDEN, HIDDEN, 0);
  gemm_xwt_kernel<<<gg, 256, 0, stream>>>(xb, wvb, vb, BS, HIDDEN, HIDDEN, 0);

  rope_kernel<<<(BS * (HIDDEN / 2)) / 256, 256, 0, stream>>>(qb, kb, fc);

  dim3 gf(SEQ / 128, BATCH * HEADS);
  flash_kernel<<<gf, 256, 0, stream>>>(qb, kb, vb, ctx);

  gemm_xwt_kernel<<<gg, 256, 0, stream>>>(ctx, wob, d_out, BS, HIDDEN, HIDDEN, 1);
}
